// LFA_55250459296222
// MI455X (gfx1250) — compile-verified, matrix operand order checked by reading, not run on hardware
//
#include <hip/hip_runtime.h>
#include <hip/hip_bf16.h>
#include <math.h>

// ---------------------------------------------------------------------------
// LFA layer for MI455X (gfx1250, wave32).
// GEMMs run on V_WMMA_F32_16X16X32_BF16 (bf16 in, fp32 accumulate).
// Everything else (gather, max, gelu, batchnorm) stays fp32.
// ---------------------------------------------------------------------------

#define N_PTS 100000
#define KNN   16
#define CIN   128
#define COUT  128
#define D2    64
#define EPSBN 1e-5f

typedef __attribute__((ext_vector_type(16))) __bf16        v16bf;
typedef __attribute__((ext_vector_type(8)))  float         v8f;
typedef __attribute__((ext_vector_type(8)))  unsigned int  v8u;

union FragU { v8u u; v16bf b; };

__device__ __forceinline__ unsigned int f2bf(float f) {
  unsigned int u = __float_as_uint(f);
  return ((u + 0x7fffu + ((u >> 16) & 1u)) >> 16) & 0xffffu;   // RNE
}
__device__ __forceinline__ unsigned int packbf(float lo, float hi) {
  return f2bf(lo) | (f2bf(hi) << 16);
}
__device__ __forceinline__ float bflo(unsigned int d) { return __uint_as_float(d << 16); }
__device__ __forceinline__ float bfhi(unsigned int d) { return __uint_as_float(d & 0xffff0000u); }

// A fragment (16x32 bf16, MxK) from LDS row-major bf16 tile.
// ISA 7.12.2: lane L holds row M=L%16; VGPR v holds K-pair
//   kpair = ks*16 + (v>=4?8:0) + (L>=16?4:0) + (v%4)   (dword units, 2 bf16/dword)
// -> v0..3 and v4..7 are each 4 consecutive dwords: two ds_read_b128 per lane.
__device__ __forceinline__ v16bf ldsA(const unsigned int* buf, int pitch,
                                      int rowBase, int ks, int lane) {
  int row  = rowBase + (lane & 15);
  int half = (lane >> 4) & 1;
  const unsigned int* p = buf + row * pitch + ks * 16 + half * 4;
  uint4 lo = *(const uint4*)(p);
  uint4 hi = *(const uint4*)(p + 8);
  FragU f;
  f.u[0]=lo.x; f.u[1]=lo.y; f.u[2]=lo.z; f.u[3]=lo.w;
  f.u[4]=hi.x; f.u[5]=hi.y; f.u[6]=hi.z; f.u[7]=hi.w;
  return f.b;
}

// B fragment (32x16 bf16, KxN) from a pre-swizzled global stream:
// dword index = (frag*32 + lane)*8 + v  -> 8 consecutive dwords per lane.
__device__ __forceinline__ v16bf glbB(const unsigned int* pk, int frag, int lane) {
  const unsigned int* p = pk + ((size_t)(frag * 32 + lane)) * 8;
  uint4 lo = *(const uint4*)(p);
  uint4 hi = *(const uint4*)(p + 4);
  FragU f;
  f.u[0]=lo.x; f.u[1]=lo.y; f.u[2]=lo.z; f.u[3]=lo.w;
  f.u[4]=hi.x; f.u[5]=hi.y; f.u[6]=hi.z; f.u[7]=hi.w;
  return f.b;
}

__device__ __forceinline__ v8f wmma_bf16(v16bf a, v16bf b, v8f c) {
  return __builtin_amdgcn_wmma_f32_16x16x32_bf16(false, a, false, b,
                                                 (short)0, c, false, false);
}

// ---------------------------------------------------------------------------
// Kernel 0: pre-swizzle fp32 weights into bf16 B-fragment streams.
// ---------------------------------------------------------------------------
__device__ void packOne(const float* w, unsigned int* dst, int Kd, int Nd,
                        int tid, int nthr) {
  int nct = Nd >> 4, nks = Kd >> 5;
  int total = nct * nks * 256;
  for (int i = tid; i < total; i += nthr) {
    int frag = i >> 8, within = i & 255;
    int lane = within >> 3, v = within & 7;
    int ct = frag / nks, ks = frag - ct * nks;
    int n  = ct * 16 + (lane & 15);
    int k  = ks * 32 + ((v >= 4) ? 16 : 0) + ((lane >= 16) ? 8 : 0) + ((v & 3) << 1);
    dst[i] = packbf(w[(size_t)k * Nd + n], w[(size_t)(k + 1) * Nd + n]);
  }
}

__global__ void pack_weights_kernel(const float* W, const float* m2w,
                                    const float* m3aw, const float* m3bw,
                                    unsigned int* wp, unsigned int* m2p,
                                    unsigned int* m3ap, unsigned int* m3bp) {
  int tid = threadIdx.x;
  packOne(W,    wp,   CIN, COUT, tid, 256);
  packOne(m2w,  m2p,  D2,  D2,   tid, 256);
  packOne(m3aw, m3ap, COUT, COUT, tid, 256);
  packOne(m3bw, m3bp, COUT, COUT, tid, 256);
}

// ---------------------------------------------------------------------------
// Kernel 1: xp = x @ W   (N x 128 @ 128 x 128), fp32 out (stays L2-resident).
// Block = 128 threads (4 waves), 32 rows per block. LDS x-tile pitch 68 dw
// (rows 16B aligned, odd-ish bank rotation).
// ---------------------------------------------------------------------------
__global__ void proj_kernel(const float* __restrict__ x,
                            const unsigned int* __restrict__ wp,
                            float* __restrict__ xp) {
  extern __shared__ unsigned int sm[];           // 32 * 68 dwords
  const int tid   = threadIdx.x;
  const int rows0 = blockIdx.x * 32;
#pragma unroll
  for (int i = 0; i < 16; ++i) {                 // stage 32x128 fp32 -> bf16
    int dw = tid + i * 128;                      // 0..2047
    int row = dw >> 6, c = dw & 63;
    const float* src = x + (size_t)(rows0 + row) * CIN + c * 2;
    sm[row * 68 + c] = packbf(src[0], src[1]);
  }
  __syncthreads();
  const int lane = tid & 31, wv = tid >> 5;
#pragma unroll
  for (int t = 0; t < 4; ++t) {
    int tile = wv * 4 + t;
    int rt = tile >> 3, ct = tile & 7;
    v8f acc = {};
#pragma unroll
    for (int ks = 0; ks < 4; ++ks)
      acc = wmma_bf16(ldsA(sm, 68, rt * 16, ks, lane), glbB(wp, ct * 4 + ks, lane), acc);
    int col = ct * 16 + (lane & 15);
    int half = (lane >> 4) & 1;
#pragma unroll
    for (int j = 0; j < 8; ++j) {                // C/D layout: VGPR j -> M=j / j+8
      int row = rows0 + rt * 16 + j + half * 8;
      xp[(size_t)row * COUT + col] = acc[j];
    }
  }
}

// ---------------------------------------------------------------------------
// Kernel 2: fused NCA. One workgroup = 4 points = 64 neighbor rows,
// 128 threads = 4 waves; wave w owns row-tile / point w.
// LDS: sIdx(64) | p0 64x36 | pfeat 64x68 | h 64x68  = 11072 dw = 44.3 KB.
// ---------------------------------------------------------------------------
#define P0_PITCH 36
#define PF_PITCH 68
#define SM_SIDX  0
#define SM_P0    64
#define SM_PF    (64 + 64 * P0_PITCH)                    // 2368
#define SM_HB    (SM_PF + 64 * PF_PITCH)                 // 6720
#define SM_TOTAL (SM_HB + 64 * PF_PITCH)                 // 11072 dwords

__global__ void fused_kernel(const float* __restrict__ xyz,
                             const int*   __restrict__ knn,
                             const float* __restrict__ xp,
                             const float* __restrict__ m1w, const float* __restrict__ m1b,
                             const unsigned int* __restrict__ m2p,  const float* __restrict__ m2b,
                             const unsigned int* __restrict__ m3ap, const float* __restrict__ m3ab,
                             const unsigned int* __restrict__ m3bp, const float* __restrict__ m3bb,
                             float* __restrict__ out_raw) {
  extern __shared__ unsigned int sm[];
  const int tid  = threadIdx.x;                  // 0..127
  const int base = blockIdx.x * 4;               // first point of this block

  // ---- phase 0: xyz deltas + p0 = xyzs@m1 + b (VALU; K-dim is only 3) ----
  {
    int r = tid >> 1;                            // row 0..63  (point*16 + k)
    int halfc = tid & 1;
    int p = r >> 4, k = r & 15;
    int gp = base + p;
    int idx = knn[(size_t)gp * KNN + k];
    if (halfc == 0) sm[SM_SIDX + r] = (unsigned int)idx;
    float cx = xyz[(size_t)gp * 3], cy = xyz[(size_t)gp * 3 + 1], cz = xyz[(size_t)gp * 3 + 2];
    float dx = xyz[(size_t)idx * 3]     - cx;
    float dy = xyz[(size_t)idx * 3 + 1] - cy;
    float dz = xyz[(size_t)idx * 3 + 2] - cz;
    int c0 = halfc * 16;                         // dword col base (2 bf16/dword)
#pragma unroll
    for (int c = 0; c < 16; ++c) {
      int j0 = (c0 + c) * 2;
      float lo = m1b[j0]     + dx * m1w[j0]     + dy * m1w[D2 + j0]     + dz * m1w[2 * D2 + j0];
      float hi = m1b[j0 + 1] + dx * m1w[j0 + 1] + dy * m1w[D2 + j0 + 1] + dz * m1w[2 * D2 + j0 + 1];
      sm[SM_P0 + r * P0_PITCH + c0 + c] = packbf(lo, hi);
    }
  }
  __syncthreads();

  const int lane = tid & 31, wv = tid >> 5;      // wv = row-tile = point-in-block

  // ---- phase 1: p_local = max_k p0  -> pfeat cols 64..127 (broadcast) ----
  {
    int p = tid >> 5, c = tid & 31;              // 4 pts x 32 dword-cols
    float mlo = -INFINITY, mhi = -INFINITY;
#pragma unroll
    for (int k = 0; k < 16; ++k) {
      unsigned int d = sm[SM_P0 + (p * 16 + k) * P0_PITCH + c];
      mlo = fmaxf(mlo, bflo(d)); mhi = fmaxf(mhi, bfhi(d));
    }
    unsigned int pk = (__float_as_uint(mhi) & 0xffff0000u) | (__float_as_uint(mlo) >> 16);
#pragma unroll
    for (int k = 0; k < 16; ++k)
      sm[SM_PF + (p * 16 + k) * PF_PITCH + 32 + c] = pk;
  }

  // ---- phase 2: p1 = p0 @ m2 + b  -> pfeat cols 0..63 (WMMA) ----
  {
    unsigned short* s16 = (unsigned short*)sm;
#pragma unroll
    for (int ct = 0; ct < 4; ++ct) {
      v8f acc = {};
#pragma unroll
      for (int ks = 0; ks < 2; ++ks)
        acc = wmma_bf16(ldsA(sm + SM_P0, P0_PITCH, wv * 16, ks, lane),
                        glbB(m2p, ct * 2 + ks, lane), acc);
      int col = ct * 16 + (lane & 15);
      int half = (lane >> 4) & 1;
      float bias = m2b[col];
#pragma unroll
      for (int j = 0; j < 8; ++j) {
        int row = wv * 16 + j + half * 8;
        s16[(SM_PF + row * PF_PITCH) * 2 + col] = (unsigned short)f2bf(acc[j] + bias);
      }
    }
  }
  __syncthreads();

  // ---- phase 3: h = gelu(pfeat @ m3a + b) -> hbuf (WMMA + exact erf gelu) ----
  {
    unsigned short* s16 = (unsigned short*)sm;
    v16bf aF[4];
#pragma unroll
    for (int ks = 0; ks < 4; ++ks) aF[ks] = ldsA(sm + SM_PF, PF_PITCH, wv * 16, ks, lane);
#pragma unroll
    for (int ct = 0; ct < 8; ++ct) {
      v8f acc = {};
#pragma unroll
      for (int ks = 0; ks < 4; ++ks)
        acc = wmma_bf16(aF[ks], glbB(m3ap, ct * 4 + ks, lane), acc);
      int col = ct * 16 + (lane & 15);
      int half = (lane >> 4) & 1;
      float bias = m3ab[col];
#pragma unroll
      for (int j = 0; j < 8; ++j) {
        float v = acc[j] + bias;
        v = 0.5f * v * (1.0f + erff(v * 0.70710678118654752f));
        s16[(SM_HB + (wv * 16 + j + half * 8) * PF_PITCH) * 2 + col] = (unsigned short)f2bf(v);
      }
    }
  }
  __syncthreads();

  // ---- phase 4: pe = h @ m3b + b; add gathered xs; max over k -> out_raw ----
  {
    v16bf aF[4];
#pragma unroll
    for (int ks = 0; ks < 4; ++ks) aF[ks] = ldsA(sm + SM_HB, PF_PITCH, wv * 16, ks, lane);
    int gp = base + wv;
#pragma unroll
    for (int ct = 0; ct < 8; ++ct) {
      v8f acc = {};
#pragma unroll
      for (int ks = 0; ks < 4; ++ks)
        acc = wmma_bf16(aF[ks], glbB(m3bp, ct * 4 + ks, lane), acc);
      int col = ct * 16 + (lane & 15);
      int half = (lane >> 4) & 1;
      float bias = m3bb[col];
      float xc = xp[(size_t)gp * COUT + col];    // L2-resident gather base
      float m = -INFINITY;
#pragma unroll
      for (int j = 0; j < 8; ++j) {              // rows of tile = neighbor index k
        int kk = wv * 16 + j + half * 8;
        unsigned int idx = sm[SM_SIDX + kk];
        float nb = xp[(size_t)idx * COUT + col];
        m = fmaxf(m, acc[j] + bias + nb - xc);
      }
      m = fmaxf(m, __shfl_xor(m, 16, 32));       // combine lane-halves (k and k+8)
      if (half == 0) out_raw[(size_t)gp * COUT + col] = m;
    }
  }
}

// ---------------------------------------------------------------------------
// BatchNorm: deterministic two-stage column reduction + in-place apply.
// ---------------------------------------------------------------------------
#define BN_BLOCKS 512

__global__ void bn_partial(const float* __restrict__ raw, float* __restrict__ partial) {
  int c = threadIdx.x;                            // 128 threads = one column each
  float s = 0.f, ss = 0.f;
  for (int r = blockIdx.x; r < N_PTS; r += gridDim.x) {
    float v = raw[(size_t)r * COUT + c];
    s += v; ss += v * v;
  }
  partial[(size_t)blockIdx.x * 256 + c]       = s;
  partial[(size_t)blockIdx.x * 256 + 128 + c] = ss;
}

__global__ void bn_combine(const float* __restrict__ partial, float* __restrict__ stats) {
  int c = threadIdx.x;
  float s = 0.f, ss = 0.f;
  for (int b = 0; b < BN_BLOCKS; ++b) {
    s  += partial[(size_t)b * 256 + c];
    ss += partial[(size_t)b * 256 + 128 + c];
  }
  float mean = s * (1.0f / (float)N_PTS);
  float var  = ss * (1.0f / (float)N_PTS) - mean * mean;
  stats[c]       = mean;
  stats[128 + c] = 1.0f / sqrtf(var + EPSBN);
}

__global__ void bn_apply(float* __restrict__ out, const float* __restrict__ stats,
                         const float* __restrict__ gamma, const float* __restrict__ beta) {
  size_t i = (size_t)blockIdx.x * blockDim.x + threadIdx.x;
  if (i < (size_t)N_PTS * COUT) {
    int c = (int)(i & (COUT - 1));
    out[i] = (out[i] - stats[c]) * stats[128 + c] * gamma[c] + beta[c];
  }
}

// ---------------------------------------------------------------------------
// Launch. Workspace layout (floats):
//   xp: N*128 (51.2 MB) | packed weights (26624 dw) | partials | stats
// ---------------------------------------------------------------------------
extern "C" void kernel_launch(void* const* d_in, const int* in_sizes, int n_in,
                              void* d_out, int out_size, void* d_ws, size_t ws_size,
                              hipStream_t stream) {
  const float* xyz  = (const float*)d_in[0];
  const float* x    = (const float*)d_in[1];
  const int*   knn  = (const int*)  d_in[2];
  const float* W    = (const float*)d_in[3];
  const float* m1w  = (const float*)d_in[4];
  const float* m1b  = (const float*)d_in[5];
  const float* m2w  = (const float*)d_in[6];
  const float* m2b  = (const float*)d_in[7];
  const float* m3aw = (const float*)d_in[8];
  const float* m3ab = (const float*)d_in[9];
  const float* m3bw = (const float*)d_in[10];
  const float* m3bb = (const float*)d_in[11];
  const float* gamma= (const float*)d_in[12];
  const float* beta = (const float*)d_in[13];
  float* out = (float*)d_out;

  float* xp = (float*)d_ws;
  unsigned int* wp   = (unsigned int*)(xp + (size_t)N_PTS * COUT);
  unsigned int* m2p  = wp   + 8192;
  unsigned int* m3ap = m2p  + 2048;
  unsigned int* m3bp = m3ap + 8192;
  float* partial = (float*)(m3bp + 8192);
  float* stats   = partial + (size_t)BN_BLOCKS * 256;

  pack_weights_kernel<<<1, 256, 0, stream>>>(W, m2w, m3aw, m3bw, wp, m2p, m3ap, m3bp);
  proj_kernel<<<N_PTS / 32, 128, 32 * 68 * 4, stream>>>(x, wp, xp);
  fused_kernel<<<N_PTS / 4, 128, SM_TOTAL * 4, stream>>>(
      xyz, knn, xp, m1w, m1b, m2p, m2b, m3ap, m3ab, m3bp, m3bb, out);
  bn_partial<<<BN_BLOCKS, 128, 0, stream>>>(out, partial);
  bn_combine<<<1, 128, 0, stream>>>(partial, stats);
  bn_apply<<<(N_PTS * COUT + 255) / 256, 256, 0, stream>>>(out, stats, gamma, beta);
}
